// WindowAttention3D_37194416783490
// MI455X (gfx1250) — matrix operand order repaired and checked by reading, not compiled, not run on hardware
//
#include <hip/hip_runtime.h>

typedef _Float16 f16;
typedef _Float16 v16h __attribute__((ext_vector_type(16)));
typedef _Float16 v8h  __attribute__((ext_vector_type(8)));
typedef float    v8f  __attribute__((ext_vector_type(8)));

#define DIM    192
#define HEADS  6
#define HD     32
#define NTOK   98
#define MPAD   112          // 7 * 16 token tiles
#define XP     200          // x / attn-out pitch (halfs)  -> 400B rows, 16B aligned
#define QP     40           // q/k pitch (halfs)           -> 80B rows
#define VP     136          // vT pitch (tokens, halfs)    -> 272B rows
#define LP     132          // logits pitch (floats)
#define PP     136          // probs pitch (halfs)
#define NWIN   4096
#define NMASK  64
#define QSCALE 0.17677669529663689f   // 32^-0.5

// LDS layout (byte offsets), total 294144 B < 320 KB
#define XO 0                               // 112*200*2 = 44800
#define QO 44800                           // 6*112*40*2 = 53760
#define KO 98560                           // 53760
#define VO 152320                          // 6*32*136*2 = 52224
#define LO 204544                          // 112*132*4 = 59136
#define PO 263680                          // 112*136*2 = 30464
#define SMEM_BYTES 294144

// workspace layout (bytes)
#define WS_W16    0                        // (576+192)*192*2 = 294912
#define WS_BIAS6  294912                   // 6*98*98*2 = 115248
#define W16_ELEMS (4 * DIM * DIM)          // 147456

__device__ __forceinline__ v8f wmma_acc(v16h a, v16h b, v8f c) {
  return __builtin_amdgcn_wmma_f32_16x16x32_f16(false, a, false, b, (short)0, c,
                                                false, false);
}

// A fragment (16x32 f16): lane L -> row row0+(L&15); halves 0..7 = K [8*hf..+7],
// halves 8..15 = K [16+8*hf..+7], hf = L>>4.  Two 16B loads.
__device__ __forceinline__ v16h load_a(const f16* buf, int pitch, int row0,
                                       int k0, int lane) {
  int hf = lane >> 4, r = lane & 15;
  const f16* p = buf + (row0 + r) * pitch + k0 + 8 * hf;
  v8h lo = *(const v8h*)p;
  v8h hi = *(const v8h*)(p + 16);
  return __builtin_shufflevector(lo, hi, 0, 1, 2, 3, 4, 5, 6, 7, 8, 9, 10, 11,
                                 12, 13, 14, 15);
}

// B fragment (32x16 f16) from a [N][K] (K-contiguous) source: lane L -> col
// n0+(L&15), halves 0..7 = K [8*hf..+7], halves 8..15 = K [16+8*hf..+7].  Two 16B loads.
__device__ __forceinline__ v16h load_bt(const f16* buf, int pitch, int k0,
                                        int n0, int lane) {
  int hf = lane >> 4, c = lane & 15;
  const f16* p = buf + (n0 + c) * pitch + k0 + 8 * hf;
  v8h lo = *(const v8h*)p;
  v8h hi = *(const v8h*)(p + 16);
  return __builtin_shufflevector(lo, hi, 0, 1, 2, 3, 4, 5, 6, 7, 8, 9, 10, 11,
                                 12, 13, 14, 15);
}

// ---- prologue 1: convert weights f32 -> f16 into workspace -----------------
__global__ void wa3d_convert_weights(const float* __restrict__ qkv_w,
                                     const float* __restrict__ proj_w,
                                     f16* __restrict__ w16) {
  const int nq = 3 * DIM * DIM;            // 110592
  for (int i = blockIdx.x * blockDim.x + threadIdx.x; i < W16_ELEMS;
       i += gridDim.x * blockDim.x)
    w16[i] = (f16)(i < nq ? qkv_w[i] : proj_w[i - nq]);
}

// ---- prologue 2: pre-gather relative-position bias per head ----------------
// bias6[h][i][j] = rpb_table[rel_idx[i][j]][h]   (identical for every window)
__global__ void wa3d_bias6(const float* __restrict__ rpb,
                           const int* __restrict__ rel_idx,
                           f16* __restrict__ bias6) {
  const int tot = HEADS * NTOK * NTOK;     // 57624
  for (int i = blockIdx.x * blockDim.x + threadIdx.x; i < tot;
       i += gridDim.x * blockDim.x) {
    int h = i / (NTOK * NTOK);
    int ij = i - h * NTOK * NTOK;
    bias6[i] = (f16)rpb[rel_idx[ij] * HEADS + h];
  }
}

// ---- main kernel: one workgroup per window ---------------------------------
__global__ __launch_bounds__(512, 1) void wa3d_kernel(
    const float* __restrict__ x, const float* __restrict__ mask,
    const float* __restrict__ qkv_b, const float* __restrict__ proj_b,
    const f16* __restrict__ bias6, const f16* __restrict__ w16,
    float* __restrict__ out) {
  extern __shared__ char smem[];
  f16* xbuf    = (f16*)(smem + XO);   // [112][200] x, later attn-out
  f16* qbuf    = (f16*)(smem + QO);   // [6][112][40]
  f16* kbuf    = (f16*)(smem + KO);   // [6][112][40]
  f16* vtbuf   = (f16*)(smem + VO);   // [6][32][136]  (v transposed)
  float* logit = (float*)(smem + LO); // [112][132]
  f16* probs   = (f16*)(smem + PO);   // [112][136]

  const int b    = blockIdx.x;
  const int tid  = threadIdx.x;
  const int lane = tid & 31;
  const int wid  = tid >> 5;
  const int NW   = 16;                       // 512/32 waves

  const f16* wqkv  = w16;                    // [576][192]
  const f16* wproj = w16 + 3 * DIM * DIM;    // [192][192]
  const float* maskw = mask + (size_t)(b & (NMASK - 1)) * NTOK * NTOK;
  const float* xg = x + (size_t)b * NTOK * DIM;

  // warm L2/L0 with the f16 weights (global_prefetch_b8); 512 threads cover
  // the 294912B weight array in 128B strides.
  for (int i = tid * 128; i < W16_ELEMS * 2; i += 512 * 128)
    __builtin_prefetch((const char*)w16 + i, 0, 3);

  // ---- phase 0: stage x (f32->f16, zero-pad rows), zero vT & probs pads ----
  for (int i = tid; i < MPAD * XP; i += 512) {
    int r = i / XP, c = i - r * XP;
    float v = (r < NTOK && c < DIM) ? xg[r * DIM + c] : 0.0f;
    xbuf[i] = (f16)v;
  }
  for (int i = tid; i < HEADS * HD * VP; i += 512) vtbuf[i] = (f16)0.0f;
  for (int i = tid; i < MPAD * PP; i += 512) probs[i] = (f16)0.0f;
  __syncthreads();

  // ---- phase 1: QKV GEMM  (112x576 = 7x36 tiles, K = 192 = 6 wmma steps) ---
  for (int t = wid; t < 7 * 36; t += NW) {
    int mt = t / 36, nt = t - mt * 36;
    int m0 = mt * 16, n0 = nt * 16;
    v8f acc = {0, 0, 0, 0, 0, 0, 0, 0};
#pragma unroll
    for (int ks = 0; ks < 6; ++ks) {
      v16h a  = load_a(xbuf, XP, m0, ks * 32, lane);
      v16h bf = load_bt(wqkv, DIM, ks * 32, n0, lane);
      acc = wmma_acc(a, bf, acc);
    }
    int col = n0 + (lane & 15);
    int r0  = (lane >> 4) * 8;         // C layout: lane holds rows r0..r0+7
    int which = col / DIM;             // uniform per tile: 0=q 1=k 2=v
    int c = col - which * DIM;
    int head = c >> 5, hd = c & 31;
    float bias = qkv_b[col];
    if (which == 0) {
#pragma unroll
      for (int vi = 0; vi < 8; ++vi)
        qbuf[head * MPAD * QP + (m0 + r0 + vi) * QP + hd] =
            (f16)((acc[vi] + bias) * QSCALE);
    } else if (which == 1) {
#pragma unroll
      for (int vi = 0; vi < 8; ++vi)
        kbuf[head * MPAD * QP + (m0 + r0 + vi) * QP + hd] =
            (f16)(acc[vi] + bias);
    } else {  // v: lane's 8 rows are consecutive tokens -> one ds_store_b128
      v8h pk;
#pragma unroll
      for (int vi = 0; vi < 8; ++vi) pk[vi] = (f16)(acc[vi] + bias);
      *(v8h*)&vtbuf[head * HD * VP + hd * VP + m0 + r0] = pk;
    }
  }
  __syncthreads();

  // ---- phase 2: per-head attention ----------------------------------------
  for (int head = 0; head < HEADS; ++head) {
    const f16* qh  = qbuf + head * MPAD * QP;
    const f16* kh  = kbuf + head * MPAD * QP;
    const f16* vh  = vtbuf + head * HD * VP;
    const f16* b6h = bias6 + head * NTOK * NTOK;

    // 2a: logits = q @ k^T  (+ pre-gathered bias + window mask), K = 32 = hd
    for (int t = wid; t < 49; t += NW) {
      int it = t / 7, jt = t - it * 7;
      v16h a  = load_a(qh, QP, it * 16, 0, lane);
      v16h bf = load_bt(kh, QP, 0, jt * 16, lane);
      v8f acc = {0, 0, 0, 0, 0, 0, 0, 0};
      acc = wmma_acc(a, bf, acc);
      int j  = jt * 16 + (lane & 15);
      int r0 = (lane >> 4) * 8;
#pragma unroll
      for (int vi = 0; vi < 8; ++vi) {
        int i = it * 16 + r0 + vi;
        float v;
        if (j < NTOK) {
          if (i < NTOK)
            v = acc[vi] + (float)b6h[i * NTOK + j] + maskw[i * NTOK + j];
          else
            v = 0.0f;                  // padded query row: keep finite
        } else {
          v = -1e30f;                  // padded key: prob -> 0
        }
        logit[i * LP + j] = v;
      }
    }
    __syncthreads();

    // 2b: row softmax (f32), 4 threads per row, shuffle reductions
    {
      int row = tid >> 2;
      int sub = tid & 3;
      if (row < MPAD) {
        float m = -1e30f;
        for (int j = sub; j < MPAD; j += 4)
          m = fmaxf(m, logit[row * LP + j]);
        m = fmaxf(m, __shfl_xor(m, 1, 32));
        m = fmaxf(m, __shfl_xor(m, 2, 32));
        float s = 0.0f;
        for (int j = sub; j < MPAD; j += 4) {
          float e = __expf(logit[row * LP + j] - m);
          logit[row * LP + j] = e;
          s += e;
        }
        s += __shfl_xor(s, 1, 32);
        s += __shfl_xor(s, 2, 32);
        float inv = 1.0f / s;
        for (int j = sub; j < MPAD; j += 4)
          probs[row * PP + j] = (f16)(logit[row * LP + j] * inv);
      }
    }
    __syncthreads();

    // 2c: out = probs @ v  (7x2 tiles, K = 128 padded tokens = 4 wmma steps)
    for (int t = wid; t < 14; t += NW) {
      int nt = t / 7, it = t - nt * 7;
      v8f acc = {0, 0, 0, 0, 0, 0, 0, 0};
#pragma unroll
      for (int kb = 0; kb < 4; ++kb) {
        v16h a  = load_a(probs, PP, it * 16, kb * 32, lane);
        v16h bf = load_bt(vh, VP, kb * 32, nt * 16, lane);
        acc = wmma_acc(a, bf, acc);
      }
      int col = head * HD + nt * 16 + (lane & 15);
      int r0  = (lane >> 4) * 8;
#pragma unroll
      for (int vi = 0; vi < 8; ++vi)
        xbuf[(it * 16 + r0 + vi) * XP + col] = (f16)acc[vi];
    }
    __syncthreads();
  }

  // ---- phase 3: output projection (7x12 tiles, K = 192 = 6 wmma steps) ----
  float* og = out + (size_t)b * NTOK * DIM;
  for (int t = wid; t < 7 * 12; t += NW) {
    int mt = t / 12, nt = t - mt * 12;
    int m0 = mt * 16, n0 = nt * 16;
    v8f acc = {0, 0, 0, 0, 0, 0, 0, 0};
#pragma unroll
    for (int ks = 0; ks < 6; ++ks) {
      v16h a  = load_a(xbuf, XP, m0, ks * 32, lane);
      v16h bf = load_bt(wproj, DIM, ks * 32, n0, lane);
      acc = wmma_acc(a, bf, acc);
    }
    int col = n0 + (lane & 15);
    int r0  = (lane >> 4) * 8;
    float pb = proj_b[col];
#pragma unroll
    for (int vi = 0; vi < 8; ++vi) {
      int i = m0 + r0 + vi;
      if (i < NTOK) og[i * DIM + col] = acc[vi] + pb;
    }
  }
}

extern "C" void kernel_launch(void* const* d_in, const int* in_sizes, int n_in,
                              void* d_out, int out_size, void* d_ws,
                              size_t ws_size, hipStream_t stream) {
  (void)in_sizes; (void)n_in; (void)out_size; (void)ws_size;
  const float* x      = (const float*)d_in[0];
  const float* mask   = (const float*)d_in[1];
  const float* qkv_w  = (const float*)d_in[2];
  const float* qkv_b  = (const float*)d_in[3];
  const float* proj_w = (const float*)d_in[4];
  const float* proj_b = (const float*)d_in[5];
  const float* rpb    = (const float*)d_in[6];
  const int*   rel    = (const int*)d_in[7];

  f16* w16   = (f16*)((char*)d_ws + WS_W16);    // 294912 B converted weights
  f16* bias6 = (f16*)((char*)d_ws + WS_BIAS6);  // 115248 B pre-gathered bias

  wa3d_convert_weights<<<dim3(288), dim3(512), 0, stream>>>(qkv_w, proj_w, w16);
  wa3d_bias6<<<dim3(113), dim3(512), 0, stream>>>(rpb, rel, bias6);

  (void)hipFuncSetAttribute(reinterpret_cast<const void*>(wa3d_kernel),
                            hipFuncAttributeMaxDynamicSharedMemorySize,
                            SMEM_BYTES);
  wa3d_kernel<<<dim3(NWIN), dim3(512), SMEM_BYTES, stream>>>(
      x, mask, qkv_b, proj_b, bias6, w16, (float*)d_out);
}
